// Lista_44573170598320
// MI455X (gfx1250) — matrix operand order repaired and checked by reading, not compiled, not run on hardware
//
#include <hip/hip_runtime.h>
#include <hip/hip_bf16.h>

// ---------------------------------------------------------------------------
// LISTA denoiser, fused, for MI455X (gfx1250, wave32, WMMA).
//
// Roofline: ~172 GFLOP of 243x256 matvecs vs ~2.3 MB unique HBM traffic
// (~0.1us @ 23.3 TB/s) => matrix-engine bound. Strategy:
//  * batch 57,600 spatial positions as GEMM-N, tile NT=32 per workgroup
//  * v_wmma_f32_16x16x32_bf16 (bf16 in, f32 accum), K padded 243->256
//  * weights A/Dw staged once into LDS (~270KB of the 320KB WGP LDS),
//    reused across all 12 unfoldings; gamma stays in VGPR accumulators
//  * 264-element LDS pitch => conflict-free ds_load_b128 fragment reads
//  * col2im via atomicAdd scatter + analytic overlap-count divide
// ---------------------------------------------------------------------------

#define PITCH 264      // LDS row/col pitch in bf16 elems (16B aligned, bank-friendly)
#define NT    32       // spatial positions per workgroup
#define FDIM  256      // dictionary atoms
#define CKK   243      // 3*9*9 patch dim
#define KS    9
#define UNF   12
#define HO    120
#define WO    120
#define HH    128
#define WPIX  128
#define BATCH 4
#define CHN   3

typedef __attribute__((ext_vector_type(16))) __bf16 v16bf;
typedef __attribute__((ext_vector_type(8)))  float  v8f;

union FragU { v16bf v; uint4 u[2]; };

// A operand: LDS row-major [row][k], pitch PITCH. ISA 16-bit A layout:
// lane m(0..15): K 0-7 (u[0]) and 16-23 (u[1]); lane m+16: K 8-15 and 24-31.
__device__ __forceinline__ v16bf load_afrag(const __bf16* base, int mrow,
                                            int kc, int lane) {
    const int half = lane >> 4;
    const __bf16* p = base + (mrow + (lane & 15)) * PITCH + kc * 32 + half * 8;
    FragU f;
    f.u[0] = *(const uint4*)p;
    f.u[1] = *(const uint4*)(p + 16);
    return f.v;
}

// B operand: LDS column-major [col][k], pitch PITCH. ISA 16-bit B layout:
// lane n(0..15): K 0-7 (u[0]) and 16-23 (u[1]); lane n+16: K 8-15 and 24-31.
__device__ __forceinline__ v16bf load_bfrag(const __bf16* base, int ncol,
                                            int kc, int lane) {
    const int half = lane >> 4;
    const __bf16* p = base + (ncol + (lane & 15)) * PITCH + kc * 32 + half * 8;
    FragU f;
    f.u[0] = *(const uint4*)p;
    f.u[1] = *(const uint4*)(p + 16);
    return f.v;
}

// acc[2][2] covers rows [mbase, mbase+32) x cols [0, 32), K = 256 (8 chunks).
__device__ __forceinline__ void mm_step(const __bf16* Abase, int mbase,
                                        const __bf16* Bbase,
                                        v8f (&acc)[2][2], int lane) {
#pragma unroll
    for (int kc = 0; kc < 8; ++kc) {
        v16bf a0 = load_afrag(Abase, mbase,      kc, lane);
        v16bf a1 = load_afrag(Abase, mbase + 16, kc, lane);
        v16bf b0 = load_bfrag(Bbase, 0,  kc, lane);
        v16bf b1 = load_bfrag(Bbase, 16, kc, lane);
        acc[0][0] = __builtin_amdgcn_wmma_f32_16x16x32_bf16(
            false, a0, false, b0, (short)0, acc[0][0], false, false);
        acc[0][1] = __builtin_amdgcn_wmma_f32_16x16x32_bf16(
            false, a0, false, b1, (short)0, acc[0][1], false, false);
        acc[1][0] = __builtin_amdgcn_wmma_f32_16x16x32_bf16(
            false, a1, false, b0, (short)0, acc[1][0], false, false);
        acc[1][1] = __builtin_amdgcn_wmma_f32_16x16x32_bf16(
            false, a1, false, b1, (short)0, acc[1][1], false, false);
    }
}

__device__ __forceinline__ float soft_thr(float x, float lm) {
    float t = fmaxf(fabsf(x) - lm, 0.f);
    return (x < 0.f) ? -t : t;
}

__global__ void lista_main(const float* __restrict__ Iimg,
                           const float* __restrict__ Amat,   // [256,243]
                           const float* __restrict__ Dmat,   // [243,256]
                           const float* __restrict__ Wmat,   // [243,256]
                           const float* __restrict__ lmb,    // [12,256]
                           float* __restrict__ out_acc) {
    extern __shared__ __bf16 smem[];
    __bf16* Abuf = smem;                         // 256 x PITCH (K zero-padded)
    __bf16* Dbuf = Abuf + 256 * PITCH;           // 256 x PITCH (rows>=243 zero)
    __bf16* IcB  = Dbuf + 256 * PITCH;           // NT cols x PITCH (col-major)
    __bf16* GbB  = IcB  + NT * PITCH;            // gamma bf16
    __bf16* XbB  = GbB  + NT * PITCH;            // x - I_c bf16
    float*  meanb = (float*)(XbB + NT * PITCH);  // NT floats

    const int tid  = threadIdx.x;
    const int w    = tid >> 5;
    const int lane = tid & 31;
    const int wg   = blockIdx.x;
    const int mb   = w * 32;                 // this wave's output-row base
    const int halfo = (lane >> 4) * 8;       // C/D row offset for upper lanes

    // ---- stage A (Fx243 row-major) and Dw (243x256 row-major) to bf16 LDS
    for (int i = tid; i < 256 * 256; i += 256) {
        int row = i >> 8, k = i & 255;
        float va = (k < CKK) ? Amat[row * CKK + k] : 0.f;
        Abuf[row * PITCH + k] = (__bf16)va;
        float vd = (row < CKK) ? Dmat[row * FDIM + k] : 0.f;
        Dbuf[row * PITCH + k] = (__bf16)vd;
    }

    // ---- im2col + mean-center: wave w handles columns 4w..4w+3
#pragma unroll
    for (int c4 = 0; c4 < 4; ++c4) {
        int col = w * 4 + c4;
        int p   = wg * NT + col;
        int b   = p / (HO * WO);
        int rem = p - b * (HO * WO);
        int oy  = rem / WO, ox = rem - (rem / WO) * WO;
        float vals[8];
        float s = 0.f;
#pragma unroll
        for (int t = 0; t < 8; ++t) {
            int k = lane + 32 * t;
            float v = 0.f;
            if (k < CKK) {
                int c  = k / 81, r2 = k - c * 81;
                int ky = r2 / KS, kx = r2 - ky * KS;
                v = Iimg[((b * CHN + c) * HH + oy + ky) * WPIX + ox + kx];
            }
            vals[t] = v; s += v;
        }
#pragma unroll
        for (int off = 16; off >= 1; off >>= 1) s += __shfl_xor(s, off, 32);
        float mean = s * (1.f / 243.f);
        if (lane == 0) meanb[col] = mean;
#pragma unroll
        for (int t = 0; t < 8; ++t) {
            int k = lane + 32 * t;
            if (k < CKK) IcB[col * PITCH + k] = (__bf16)(vals[t] - mean);
        }
        if (lane < 21) IcB[col * PITCH + CKK + lane] = (__bf16)0.f;  // zero pad
    }
    __syncthreads();

    // ---- initial lin = A @ I_c, then soft-threshold with lmbdas[0]
    v8f g[2][2];
#pragma unroll
    for (int mi = 0; mi < 2; ++mi)
#pragma unroll
        for (int ni = 0; ni < 2; ++ni) g[mi][ni] = (v8f){0,0,0,0,0,0,0,0};
    mm_step(Abuf, mb, IcB, g, lane);
    {
        float lmr[2][8];
#pragma unroll
        for (int mi = 0; mi < 2; ++mi)
#pragma unroll
            for (int r = 0; r < 8; ++r)
                lmr[mi][r] = lmb[mb + 16 * mi + halfo + r];
#pragma unroll
        for (int mi = 0; mi < 2; ++mi)
#pragma unroll
            for (int ni = 0; ni < 2; ++ni)
#pragma unroll
                for (int r = 0; r < 8; ++r)
                    g[mi][ni][r] = soft_thr(g[mi][ni][r], lmr[mi][r]);
    }

    // ---- 11 unfoldings, gamma stays in registers
    for (int kk = 0; kk < UNF - 1; ++kk) {
        // gamma -> LDS (bf16, col-major)
#pragma unroll
        for (int mi = 0; mi < 2; ++mi)
#pragma unroll
            for (int ni = 0; ni < 2; ++ni)
#pragma unroll
                for (int r = 0; r < 8; ++r) {
                    int row = mb + 16 * mi + halfo + r;
                    int col = 16 * ni + (lane & 15);
                    GbB[col * PITCH + row] = (__bf16)g[mi][ni][r];
                }
        __syncthreads();

        // x = Dw @ gamma ; stage (x - I_c) into LDS
        v8f xa[2][2];
#pragma unroll
        for (int mi = 0; mi < 2; ++mi)
#pragma unroll
            for (int ni = 0; ni < 2; ++ni) xa[mi][ni] = (v8f){0,0,0,0,0,0,0,0};
        mm_step(Dbuf, mb, GbB, xa, lane);
#pragma unroll
        for (int mi = 0; mi < 2; ++mi)
#pragma unroll
            for (int ni = 0; ni < 2; ++ni)
#pragma unroll
                for (int r = 0; r < 8; ++r) {
                    int row = mb + 16 * mi + halfo + r;
                    int col = 16 * ni + (lane & 15);
                    float icv = (float)IcB[col * PITCH + row];
                    XbB[col * PITCH + row] = (__bf16)(xa[mi][ni][r] - icv);
                }
        __syncthreads();

        // r_k = A @ (x - I_c) ; gamma = ST(gamma - r_k, lmbdas[kk+1])
        v8f ra[2][2];
#pragma unroll
        for (int mi = 0; mi < 2; ++mi)
#pragma unroll
            for (int ni = 0; ni < 2; ++ni) ra[mi][ni] = (v8f){0,0,0,0,0,0,0,0};
        mm_step(Abuf, mb, XbB, ra, lane);
        float lmr[2][8];
#pragma unroll
        for (int mi = 0; mi < 2; ++mi)
#pragma unroll
            for (int r = 0; r < 8; ++r)
                lmr[mi][r] = lmb[(kk + 1) * FDIM + mb + 16 * mi + halfo + r];
#pragma unroll
        for (int mi = 0; mi < 2; ++mi)
#pragma unroll
            for (int ni = 0; ni < 2; ++ni)
#pragma unroll
                for (int r = 0; r < 8; ++r)
                    g[mi][ni][r] = soft_thr(g[mi][ni][r] - ra[mi][ni][r],
                                            lmr[mi][r]);
        __syncthreads();
    }

    // ---- final: out_all = Ww @ gamma + mean, scattered col2im (atomics)
#pragma unroll
    for (int mi = 0; mi < 2; ++mi)
#pragma unroll
        for (int ni = 0; ni < 2; ++ni)
#pragma unroll
            for (int r = 0; r < 8; ++r) {
                int row = mb + 16 * mi + halfo + r;
                int col = 16 * ni + (lane & 15);
                GbB[col * PITCH + row] = (__bf16)g[mi][ni][r];
            }
    __syncthreads();
    // Abuf no longer needed -> reload with Ww (all reads of A are behind the barrier)
    for (int i = tid; i < 256 * 256; i += 256) {
        int row = i >> 8, k = i & 255;
        float vw = (row < CKK) ? Wmat[row * FDIM + k] : 0.f;
        Abuf[row * PITCH + k] = (__bf16)vw;
    }
    __syncthreads();

    v8f oa[2][2];
#pragma unroll
    for (int mi = 0; mi < 2; ++mi)
#pragma unroll
        for (int ni = 0; ni < 2; ++ni) oa[mi][ni] = (v8f){0,0,0,0,0,0,0,0};
    mm_step(Abuf, mb, GbB, oa, lane);

#pragma unroll
    for (int mi = 0; mi < 2; ++mi)
#pragma unroll
        for (int ni = 0; ni < 2; ++ni)
#pragma unroll
            for (int r = 0; r < 8; ++r) {
                int cr = mb + 16 * mi + halfo + r;
                if (cr < CKK) {
                    int col = 16 * ni + (lane & 15);
                    int p   = wg * NT + col;
                    int b   = p / (HO * WO);
                    int rem = p - b * (HO * WO);
                    int oy  = rem / WO, ox = rem - (rem / WO) * WO;
                    int c   = cr / 81, r2 = cr - c * 81;
                    int ky  = r2 / KS, kx = r2 - ky * KS;
                    float v = oa[mi][ni][r] + meanb[col];
                    atomicAdd(&out_acc[((b * CHN + c) * HH + oy + ky) * WPIX
                                       + ox + kx], v);
                }
            }
}

__global__ void zero_out_k(float* __restrict__ out, int n) {
    int i = blockIdx.x * blockDim.x + threadIdx.x;
    if (i < n) out[i] = 0.f;
}

__global__ void finalize_k(float* __restrict__ out, int n) {
    int i = blockIdx.x * blockDim.x + threadIdx.x;
    if (i >= n) return;
    int wq = i % WPIX;
    int h  = (i / WPIX) % HH;
    int nh = min(h,  HO - 1) - max(0, h  - (KS - 1)) + 1;
    int nw = min(wq, WO - 1) - max(0, wq - (KS - 1)) + 1;
    out[i] *= 1.f / (float)(nh * nw);
}

extern "C" void kernel_launch(void* const* d_in, const int* in_sizes, int n_in,
                              void* d_out, int out_size, void* d_ws, size_t ws_size,
                              hipStream_t stream) {
    const float* I   = (const float*)d_in[0];  // [4,3,128,128]
    const float* A   = (const float*)d_in[1];  // [256,243]
    const float* Dw  = (const float*)d_in[2];  // [243,256]
    const float* Wwm = (const float*)d_in[3];  // [243,256]
    const float* lmb = (const float*)d_in[4];  // [12,256]
    // d_in[5] = sigma_hat: with N_SIGMA=1 lookup_arr() is constant 0 -> ignored.
    float* out = (float*)d_out;

    const int n = BATCH * CHN * HH * WPIX;     // 196608 == out_size
    zero_out_k<<<(n + 255) / 256, 256, 0, stream>>>(out, n);

    const size_t smem = (size_t)(2 * 256 * PITCH + 3 * NT * PITCH) * sizeof(__bf16)
                      + NT * sizeof(float);    // ~313.6 KB of the 320 KB WGP LDS
    const int nwg = (BATCH * HO * WO) / NT;    // 57600/32 = 1800 workgroups
    lista_main<<<nwg, 256, smem, stream>>>(I, A, Dw, Wwm, lmb, out);

    finalize_k<<<(n + 255) / 256, 256, 0, stream>>>(out, n);
}
